// SBC2Model_46514495815854
// MI455X (gfx1250) — matrix operand order repaired and checked by reading, not compiled, not run on hardware
//
#include <hip/hip_runtime.h>
#include <hip/hip_bf16.h>

// ---------------------------------------------------------------------------
// Graph-transformer forward (2 layers) for MI455X / gfx1250, wave32 + WMMA.
// Async global->LDS staging (ASYNCcnt) for transpose-free tiles.
// ---------------------------------------------------------------------------

#define NPAD   1088              // 34 * 32, padded sequence length
#define MROWS  (4 * NPAD)        // 4352 padded rows total (B=4)
#define HIDC   512
#define FFNC   2048
#define NACT   1033              // real sequence length (1024 + 9 CLS)
#define NKEY   1024
#define CLSC   9

typedef __attribute__((ext_vector_type(16))) _Float16 v16h;
typedef __attribute__((ext_vector_type(8)))  _Float16 v8h;
typedef __attribute__((ext_vector_type(8)))  float    v8f;

__device__ __forceinline__ v16h cat8(v8h lo, v8h hi) {
  return __builtin_shufflevector(lo, hi, 0,1,2,3,4,5,6,7,8,9,10,11,12,13,14,15);
}
__device__ __forceinline__ v8f wmma_f16(v16h a, v16h b, v8f c) {
  // emits v_wmma_f32_16x16x32_f16
  return __builtin_amdgcn_wmma_f32_16x16x32_f16(false, a, false, b,
                                                (short)0, c, false, false);
}

// Async 16B global -> LDS copy, tracked by ASYNCcnt (no VGPR landing zone).
__device__ __forceinline__ void async_copy_b128(void* lds_dst, const void* gsrc) {
  unsigned           ldsa = (unsigned)(uintptr_t)lds_dst;        // LDS byte offset
  unsigned long long ga   = (unsigned long long)(uintptr_t)gsrc; // 64-bit VA
  asm volatile("global_load_async_to_lds_b128 %0, %1, off"
               :: "v"(ldsa), "v"(ga) : "memory");
}
__device__ __forceinline__ void async_wait0() {
  asm volatile("s_wait_asynccnt 0x0" ::: "memory");
}

// ---------------------------------------------------------------------------
// f32 -> f16 elementwise convert (weights)
// ---------------------------------------------------------------------------
__global__ void cvt_f16_kernel(const float* __restrict__ in,
                               _Float16* __restrict__ out, int n) {
  int i = blockIdx.x * 256 + threadIdx.x;
  if (i < n) out[i] = (_Float16)in[i];
}

// ---------------------------------------------------------------------------
// Initialize padded activation buffers from node_embeds
// ---------------------------------------------------------------------------
__global__ void init_x_kernel(const float* __restrict__ node,
                              float* __restrict__ xf,
                              _Float16* __restrict__ xh) {
  int i = blockIdx.x * 256 + threadIdx.x;   // MROWS*512 total
  int row = i >> 9, c = i & 511;
  int b = row / NPAD, n = row % NPAD;
  float v = (n < NACT) ? node[((size_t)b * NACT + n) * HIDC + c] : 0.f;
  xf[i] = v;
  xh[i] = (_Float16)v;
}

// ---------------------------------------------------------------------------
// Generic WMMA GEMM: out = A(f16, MxK) @ B(f16, KxN) + bias [+resid] [relu]
// 256 threads = 8 waves. Block tile 64(M) x 128(N), wave tile 32x32,
// K-step 32. LDS: A[64][32] row-major (async-staged), B[128][32] stored [n][k].
// ---------------------------------------------------------------------------
__global__ __launch_bounds__(256)
void gemm_f16_kernel(const _Float16* __restrict__ A, const _Float16* __restrict__ B,
                     const float* __restrict__ bias, const float* __restrict__ resid,
                     float* __restrict__ outf, _Float16* __restrict__ outh,
                     int M, int N, int K, int relu) {
  __shared__ _Float16 As[64][32];
  __shared__ _Float16 Bs[128][32];

  const int tid  = threadIdx.x;
  const int m0   = blockIdx.y * 64;
  const int n0   = blockIdx.x * 128;
  const int wid  = tid >> 5, lane = tid & 31;
  const int hlf  = lane >> 4, l16 = lane & 15;
  const int wm   = wid >> 2,  wn  = wid & 3;

  const v8f vz = {0.f,0.f,0.f,0.f,0.f,0.f,0.f,0.f};
  v8f acc[2][2];
  acc[0][0] = vz; acc[0][1] = vz; acc[1][0] = vz; acc[1][1] = vz;

  const int arow = tid >> 2, ac = (tid & 3) * 8;

  for (int k0 = 0; k0 < K; k0 += 32) {
    __syncthreads();
    // A tile: 64x32 halves, one async 16B chunk per thread (no VGPR staging)
    async_copy_b128(&As[arow][ac], &A[(size_t)(m0 + arow) * K + k0 + ac]);
    // B tile: 32x128 halves, transposed into Bs[n][k] (needs VGPR pass)
    #pragma unroll
    for (int t = 0; t < 2; ++t) {
      int ci = tid + t * 256;
      int kr = ci >> 4, nc = (ci & 15) * 8;
      v8h bv = *(const v8h*)&B[(size_t)(k0 + kr) * N + n0 + nc];
      #pragma unroll
      for (int j = 0; j < 8; ++j) Bs[nc + j][kr] = bv[j];
    }
    if (k0 + 32 < K)  // warm L2 for the next A tile (global_prefetch_b8)
      __builtin_prefetch(&A[(size_t)(m0 + arow) * K + k0 + 32 + ac], 0, 1);
    async_wait0();
    __syncthreads();

    v16h af[2], bf[2];
    #pragma unroll
    for (int mt = 0; mt < 2; ++mt) {
      // A layout: row = lane%16, K runs of 8 at hlf*8 and hlf*8+16
      const _Float16* p = &As[wm * 32 + mt * 16 + l16][hlf * 8];
      af[mt] = cat8(*(const v8h*)p, *(const v8h*)(p + 16));
    }
    #pragma unroll
    for (int nt = 0; nt < 2; ++nt) {
      // B layout: col = lane%16, K runs of 8 at hlf*8 and hlf*8+16
      const _Float16* p = &Bs[wn * 32 + nt * 16 + l16][hlf * 8];
      bf[nt] = cat8(*(const v8h*)p, *(const v8h*)(p + 16));
    }
    #pragma unroll
    for (int mt = 0; mt < 2; ++mt)
      #pragma unroll
      for (int nt = 0; nt < 2; ++nt)
        acc[mt][nt] = wmma_f16(af[mt], bf[nt], acc[mt][nt]);
  }

  // Epilogue: C layout row = r + 8*hlf, col = lane%16
  #pragma unroll
  for (int mt = 0; mt < 2; ++mt)
    #pragma unroll
    for (int nt = 0; nt < 2; ++nt)
      #pragma unroll
      for (int r = 0; r < 8; ++r) {
        int row = m0 + wm * 32 + mt * 16 + r + 8 * hlf;
        int col = n0 + wn * 32 + nt * 16 + l16;
        float v = acc[mt][nt][r] + bias[col];
        if (resid) v += resid[(size_t)row * N + col];
        if (relu)  v = fmaxf(v, 0.f);
        size_t o = (size_t)row * N + col;
        if (outf) outf[o] = v;
        if (outh) outh[o] = (_Float16)v;
      }
}

// ---------------------------------------------------------------------------
// Flash attention with edge-weight gating + heat-kernel PE bias + key mask.
// Block: 128 threads = 4 waves; each wave owns a 16-row query tile.
// Per 32-key chunk: K tile async-staged to LDS, V tile transposed via VGPRs,
// S = Q K^T via 4 WMMAs, masked online softmax (shuffle row-reductions),
// P through LDS (layout transpose), O += P V via 4 WMMAs.
// ---------------------------------------------------------------------------
__global__ __launch_bounds__(128)
void attn_kernel(const _Float16* __restrict__ Q, const _Float16* __restrict__ Kh,
                 const _Float16* __restrict__ Vh, const float* __restrict__ ew,
                 const float* __restrict__ pe, const unsigned char* __restrict__ pmask,
                 _Float16* __restrict__ Oh) {
  __shared__ _Float16 kt[32][64];        // [key][dim]   (async-staged)
  __shared__ _Float16 vt[64][32];        // [dim][key]   (transposed V)
  __shared__ _Float16 pb[4][16][32];     // per-wave P tile

  const int tid = threadIdx.x;
  const int wid = tid >> 5, lane = tid & 31;
  const int hlf = lane >> 4, l16 = lane & 15;
  const int bh  = blockIdx.y, b = bh >> 3, h = bh & 7;
  const int q0  = blockIdx.x * 64 + wid * 16;

  const size_t base = ((size_t)b * NPAD) * HIDC + h * 64;

  // Q A-fragments (DK=64 -> two K=32 fragments), loaded straight from global
  v16h qf[2];
  {
    const _Float16* p = Q + base + (size_t)(q0 + l16) * HIDC + hlf * 8;
    qf[0] = cat8(*(const v8h*)p,        *(const v8h*)(p + 16));
    qf[1] = cat8(*(const v8h*)(p + 32), *(const v8h*)(p + 48));
  }

  const v8f vz = {0.f,0.f,0.f,0.f,0.f,0.f,0.f,0.f};
  v8f o[4]; o[0] = vz; o[1] = vz; o[2] = vz; o[3] = vz;
  float mr[8], lr[8];
  #pragma unroll
  for (int r = 0; r < 8; ++r) { mr[r] = -3.0e38f; lr[r] = 0.f; }

  for (int cc = 0; cc < NPAD / 32; ++cc) {
    __syncthreads();
    #pragma unroll
    for (int t = 0; t < 2; ++t) {           // 256 16B chunks over 128 threads
      int ci = tid + t * 128;
      int key = ci >> 3, d = (ci & 7) * 8;
      size_t g = base + (size_t)(cc * 32 + key) * HIDC + d;
      async_copy_b128(&kt[key][d], &Kh[g]);   // K: direct async global->LDS
      v8h vv = *(const v8h*)&Vh[g];           // V: VGPR pass for transpose
      #pragma unroll
      for (int j = 0; j < 8; ++j) vt[d + j][key] = vv[j];
    }
    async_wait0();
    __syncthreads();

    // ---- S = Q K^T for two 16-key tiles, then masked-score epilogue ----
    float sc[2][8];
    #pragma unroll
    for (int t = 0; t < 2; ++t) {
      v8f c = vz;
      #pragma unroll
      for (int ch = 0; ch < 2; ++ch) {
        const _Float16* p = &kt[t * 16 + l16][hlf * 8 + ch * 32];
        v16h kf = cat8(*(const v8h*)p, *(const v8h*)(p + 16));
        c = wmma_f16(qf[ch], kf, c);
      }
      const int key = cc * 32 + t * 16 + l16;
      #pragma unroll
      for (int r = 0; r < 8; ++r) {
        const int q = q0 + r + 8 * hlf;
        float sv = -1.0e9f;
        if (q < NACT && key < NACT) {
          float w  = ew[((size_t)b * NACT + q) * NACT + key];
          bool  km = (key < CLSC) ? true : (pmask[b * NKEY + key - CLSC] != 0);
          if (w != 0.f && km) {
            float bias = 0.f;
            if (q >= CLSC && key >= CLSC)
              bias = pe[((size_t)b * NKEY + (q - CLSC)) * NKEY + (key - CLSC)];
            sv = c[r] * 0.125f * w + bias;   // /sqrt(64) * ew + pe
          }
        }
        sc[t][r] = sv;
      }
    }

    // ---- online softmax update (row reductions across 16-lane halves) ----
    float alpha[8];
    #pragma unroll
    for (int r = 0; r < 8; ++r) {
      float mx = fmaxf(sc[0][r], sc[1][r]);
      #pragma unroll
      for (int d = 1; d < 16; d <<= 1) mx = fmaxf(mx, __shfl_xor(mx, d, 32));
      float mnew = fmaxf(mr[r], mx);
      float a  = __expf(mr[r] - mnew);
      float p0 = __expf(sc[0][r] - mnew);
      float p1 = __expf(sc[1][r] - mnew);
      float rs = p0 + p1;
      #pragma unroll
      for (int d = 1; d < 16; d <<= 1) rs += __shfl_xor(rs, d, 32);
      lr[r] = lr[r] * a + rs;
      mr[r] = mnew;
      alpha[r] = a;
      // C-layout -> LDS (row = r+8*hlf, cols l16 / 16+l16)
      pb[wid][r + 8 * hlf][l16]      = (_Float16)p0;
      pb[wid][r + 8 * hlf][16 + l16] = (_Float16)p1;
    }

    // wave-private LDS round trip: order DS store -> DS load explicitly
    asm volatile("s_wait_dscnt 0" ::: "memory");

    // P as A-fragment (16x32)
    v16h pf;
    {
      const _Float16* p = &pb[wid][l16][hlf * 8];
      pf = cat8(*(const v8h*)p, *(const v8h*)(p + 16));
    }

    // ---- O = O*alpha + P @ V ----
    #pragma unroll
    for (int nt = 0; nt < 4; ++nt) {
      v8f oo = o[nt];
      #pragma unroll
      for (int r = 0; r < 8; ++r) oo[r] *= alpha[r];
      const _Float16* p = &vt[nt * 16 + l16][hlf * 8];
      v16h vf = cat8(*(const v8h*)p, *(const v8h*)(p + 16));
      o[nt] = wmma_f16(pf, vf, oo);
    }
  }

  #pragma unroll
  for (int nt = 0; nt < 4; ++nt)
    #pragma unroll
    for (int r = 0; r < 8; ++r) {
      const int q = q0 + r + 8 * hlf;
      Oh[base + (size_t)q * HIDC + nt * 16 + l16] = (_Float16)(o[nt][r] / lr[r]);
    }
}

// ---------------------------------------------------------------------------
// Row LayerNorm over 512 columns; emits f32 (residual chain) + f16 (GEMM A).
// ---------------------------------------------------------------------------
__global__ __launch_bounds__(256)
void ln_kernel(const float* __restrict__ y, const float* __restrict__ g,
               const float* __restrict__ bb, float* __restrict__ xf,
               _Float16* __restrict__ xh) {
  __shared__ float red[16];
  const int row = blockIdx.x, tid = threadIdx.x;
  const size_t o0 = (size_t)row * HIDC + tid;
  const size_t o1 = o0 + 256;
  float v0 = y[o0], v1 = y[o1];
  float s = v0 + v1, sq = v0 * v0 + v1 * v1;
  #pragma unroll
  for (int d = 1; d < 32; d <<= 1) {
    s  += __shfl_xor(s,  d, 32);
    sq += __shfl_xor(sq, d, 32);
  }
  if ((tid & 31) == 0) { red[tid >> 5] = s; red[8 + (tid >> 5)] = sq; }
  __syncthreads();
  float ts = 0.f, tq = 0.f;
  #pragma unroll
  for (int i = 0; i < 8; ++i) { ts += red[i]; tq += red[8 + i]; }
  float mean = ts * (1.f / HIDC);
  float var  = tq * (1.f / HIDC) - mean * mean;
  float rstd = rsqrtf(var + 1e-5f);
  float r0 = (v0 - mean) * rstd * g[tid] + bb[tid];
  float r1 = (v1 - mean) * rstd * g[256 + tid] + bb[256 + tid];
  xf[o0] = r0; xf[o1] = r1;
  xh[o0] = (_Float16)r0; xh[o1] = (_Float16)r1;
}

// ---------------------------------------------------------------------------
// Strip padding into d_out (B x 1033 x 512, f32)
// ---------------------------------------------------------------------------
__global__ void copy_out_kernel(const float* __restrict__ xf,
                                float* __restrict__ out) {
  int i = blockIdx.x * 256 + threadIdx.x;      // 4*1033*512
  if (i >= 4 * NACT * HIDC) return;
  int c = i & 511, rn = i >> 9;
  int b = rn / NACT, n = rn % NACT;
  out[i] = xf[((size_t)b * NPAD + n) * HIDC + c];
}

// ---------------------------------------------------------------------------
extern "C" void kernel_launch(void* const* d_in, const int* in_sizes, int n_in,
                              void* d_out, int out_size, void* d_ws, size_t ws_size,
                              hipStream_t stream) {
  (void)in_sizes; (void)n_in; (void)out_size; (void)ws_size;

  const float* node = (const float*)d_in[0];
  const float* ew   = (const float*)d_in[1];
  const float* pe   = (const float*)d_in[2];
  const unsigned char* pmask = (const unsigned char*)d_in[3];
  const float* Wq = (const float*)d_in[4];  const float* bq = (const float*)d_in[5];
  const float* Wk = (const float*)d_in[6];  const float* bk = (const float*)d_in[7];
  const float* Wv = (const float*)d_in[8];  const float* bv = (const float*)d_in[9];
  const float* Wo = (const float*)d_in[10]; const float* bo = (const float*)d_in[11];
  const float* g1 = (const float*)d_in[12]; const float* b1 = (const float*)d_in[13];
  const float* W1 = (const float*)d_in[14]; const float* c1 = (const float*)d_in[15];
  const float* W2 = (const float*)d_in[16]; const float* c2 = (const float*)d_in[17];
  const float* g2 = (const float*)d_in[18]; const float* b2 = (const float*)d_in[19];

  // ---- carve workspace ----
  char* p = (char*)d_ws;
  auto carve = [&](size_t bytes) -> char* {
    char* r = p; p += (bytes + 255) & ~(size_t)255; return r;
  };
  _Float16* Wqh = (_Float16*)carve((size_t)512 * 512 * 2);
  _Float16* Wkh = (_Float16*)carve((size_t)512 * 512 * 2);
  _Float16* Wvh = (_Float16*)carve((size_t)512 * 512 * 2);
  _Float16* Woh = (_Float16*)carve((size_t)512 * 512 * 2);
  _Float16* W1h = (_Float16*)carve((size_t)512 * 2048 * 2);
  _Float16* W2h = (_Float16*)carve((size_t)2048 * 512 * 2);
  _Float16* xh  = (_Float16*)carve((size_t)MROWS * 512 * 2);
  _Float16* qh  = (_Float16*)carve((size_t)MROWS * 512 * 2);
  _Float16* kh  = (_Float16*)carve((size_t)MROWS * 512 * 2);
  _Float16* vh  = (_Float16*)carve((size_t)MROWS * 512 * 2);
  _Float16* ah  = (_Float16*)carve((size_t)MROWS * 512 * 2);
  _Float16* hh  = (_Float16*)carve((size_t)MROWS * 2048 * 2);
  float*    xf  = (float*)   carve((size_t)MROWS * 512 * 4);
  float*    yf  = (float*)   carve((size_t)MROWS * 512 * 4);

  // ---- weight conversion (f32 -> f16) ----
  cvt_f16_kernel<<<(512 * 512 + 255) / 256, 256, 0, stream>>>(Wq, Wqh, 512 * 512);
  cvt_f16_kernel<<<(512 * 512 + 255) / 256, 256, 0, stream>>>(Wk, Wkh, 512 * 512);
  cvt_f16_kernel<<<(512 * 512 + 255) / 256, 256, 0, stream>>>(Wv, Wvh, 512 * 512);
  cvt_f16_kernel<<<(512 * 512 + 255) / 256, 256, 0, stream>>>(Wo, Woh, 512 * 512);
  cvt_f16_kernel<<<(512 * 2048 + 255) / 256, 256, 0, stream>>>(W1, W1h, 512 * 2048);
  cvt_f16_kernel<<<(2048 * 512 + 255) / 256, 256, 0, stream>>>(W2, W2h, 2048 * 512);

  // ---- init padded activations ----
  init_x_kernel<<<(MROWS * 512) / 256, 256, 0, stream>>>(node, xf, xh);

  const dim3 gemm512(512 / 128, MROWS / 64);    // (4, 68)
  const dim3 gemmFFN(FFNC / 128, MROWS / 64);   // (16, 68)
  const dim3 attnG(NPAD / 64, 4 * 8);           // (17, 32)

  for (int layer = 0; layer < 2; ++layer) {
    // QKV projections
    gemm_f16_kernel<<<gemm512, 256, 0, stream>>>(xh, Wqh, bq, nullptr, nullptr, qh,
                                                 MROWS, 512, 512, 0);
    gemm_f16_kernel<<<gemm512, 256, 0, stream>>>(xh, Wkh, bk, nullptr, nullptr, kh,
                                                 MROWS, 512, 512, 0);
    gemm_f16_kernel<<<gemm512, 256, 0, stream>>>(xh, Wvh, bv, nullptr, nullptr, vh,
                                                 MROWS, 512, 512, 0);
    // gated/biased flash attention
    attn_kernel<<<attnG, 128, 0, stream>>>(qh, kh, vh, ew, pe, pmask, ah);
    // output projection + residual, then LN1
    gemm_f16_kernel<<<gemm512, 256, 0, stream>>>(ah, Woh, bo, xf, yf, nullptr,
                                                 MROWS, 512, 512, 0);
    ln_kernel<<<MROWS, 256, 0, stream>>>(yf, g1, b1, xf, xh);
    // FFN
    gemm_f16_kernel<<<gemmFFN, 256, 0, stream>>>(xh, W1h, c1, nullptr, nullptr, hh,
                                                 MROWS, FFNC, 512, 1);
    gemm_f16_kernel<<<gemm512, 256, 0, stream>>>(hh, W2h, c2, xf, yf, nullptr,
                                                 MROWS, 512, FFNC, 0);
    ln_kernel<<<MROWS, 256, 0, stream>>>(yf, g2, b2, xf, xh);
  }

  copy_out_kernel<<<(4 * NACT * HIDC + 255) / 256, 256, 0, stream>>>(xf, (float*)d_out);
}
